// FlatSwiGLUS2Merge_78993038508791
// MI455X (gfx1250) — matrix operand order repaired and checked by reading, not compiled, not run on hardware
//
#include <hip/hip_runtime.h>

typedef _Float16 h2   __attribute__((ext_vector_type(2)));
typedef _Float16 h8   __attribute__((ext_vector_type(8)));
typedef _Float16 v16h __attribute__((ext_vector_type(16)));
typedef float    v8f  __attribute__((ext_vector_type(8)));

#define CC     128          // C
#define JJ     25           // (LMAX+1)^2
#define AA     196          // 14*14
#define JPAD   32           // K pad for GEMM1
#define APAD   224          // K pad for GEMM2 (7*32)
#define AMPAD  208          // M pad for GEMM1 (13*16)
#define NROW   6528         // C + 2*C*J
#define MAINW  6400         // 2*C*J
#define OUTROW 3200         // C*J

// ---- LDS layout (bytes), regions aliased where lifetimes don't overlap ----
#define OFF_T   0           // to_grid f16   208*32*2 = 13312
#define OFF_F   13312       // from_grid f16  32*224*2 = 14336
#define OFF_XT  27648       // x_arr^T f16   256*32*2 = 16384  (later: out f32 25*128*4=12800)
#define OFF_PT  44032       // prod^T  f16   128*224*2 = 57344 (earlier: raw row f32 6400*4=25600)
#define OFF_SIG 101376      // sigmoid(gate) f32 128*4 = 512
#define OFF_OSC 101888      // swiglu scalars f32 128*4 = 512
#define LDS_BYTES 102400

// A-fragment (16x32 f16, M x K): lane = row (0..15); lanes 0-15 hold K 0-7 & 16-23,
// lanes 16-31 hold K 8-15 & 24-31 (per ISA 16-bit A 16x32 table).
static __device__ inline v16h load_afrag(const _Float16* base, int strideH, int kOff, int lane) {
    int r  = lane & 15;
    int kb = kOff + ((lane >> 4) << 3);           // 0 or 8
    const _Float16* p = base + r * strideH + kb;
    h8 lo = *(const h8*)(p);                      // K kb .. kb+7
    h8 hi = *(const h8*)(p + 16);                 // K kb+16 .. kb+23
    return __builtin_shufflevector(lo, hi, 0,1,2,3,4,5,6,7,8,9,10,11,12,13,14,15);
}

// B-fragment (32x16 f16, K x N), stored column-major (row c of LDS = column of B):
// lane n (0-15) holds K 0-7 & 16-23 of column n, lane 16+n holds K 8-15 & 24-31 (matches A).
static __device__ inline v16h load_bfrag(const _Float16* base, int strideH, int kOff, int lane) {
    int nn = lane & 15;
    int kb = kOff + ((lane >> 4) << 3);           // 0 or 8
    const _Float16* p = base + nn * strideH + kb;
    h8 lo = *(const h8*)(p);                      // K kb .. kb+7
    h8 hi = *(const h8*)(p + 16);                 // K kb+16 .. kb+23
    return __builtin_shufflevector(lo, hi, 0,1,2,3,4,5,6,7,8,9,10,11,12,13,14,15);
}

__global__ __launch_bounds__(256)
void s2_swiglu_kernel(const float* __restrict__ x_flat,
                      const float* __restrict__ to_grid,
                      const float* __restrict__ from_grid,
                      const int*   __restrict__ pack_in,
                      const int*   __restrict__ unpack_out,
                      float*       __restrict__ out)
{
    __shared__ __align__(16) unsigned char smem[LDS_BYTES];
    _Float16* Tl   = (_Float16*)(smem + OFF_T);
    _Float16* Fl   = (_Float16*)(smem + OFF_F);
    _Float16* XT   = (_Float16*)(smem + OFF_XT);
    float*    OutL = (float*)   (smem + OFF_XT);   // alias: live after XT is dead
    _Float16* PT   = (_Float16*)(smem + OFF_PT);
    float*    Raw  = (float*)   (smem + OFF_PT);   // alias: live before PT is written
    float*    Sig  = (float*)   (smem + OFF_SIG);
    float*    Osc  = (float*)   (smem + OFF_OSC);

    const int tid  = threadIdx.x;
    const int lane = tid & 31;
    const int wave = tid >> 5;                     // 0..7
    const int n    = blockIdx.x;
    const float* row   = x_flat + (size_t)n * NROW;
    const float* mainp = row + CC;

    // ---------------- Phase 0: stage operands ----------------
    for (int e = tid; e < MAINW; e += 256) Raw[e] = mainp[e];           // coalesced row load
    for (int i = tid; i < AMPAD * JPAD; i += 256) {                     // to_grid -> f16, zero-pad
        int a = i >> 5, j = i & 31;
        float v = (a < AA && j < JJ) ? to_grid[a * JJ + j] : 0.f;
        Tl[i] = (_Float16)v;
    }
    for (int i = tid; i < 32 * APAD; i += 256) {                        // from_grid -> f16, zero-pad
        int j = i / APAD, a = i - j * APAD;
        float v = (j < JJ && a < AA) ? from_grid[j * AA + a] : 0.f;
        Fl[i] = (_Float16)v;
    }
    for (int i = tid; i < (256 * JPAD) / 2; i += 256)                   // zero XT (K-pad rows)
        ((unsigned int*)XT)[i] = 0u;
    if (tid < CC) {
        float g = row[tid];
        Sig[tid] = 1.f / (1.f + __expf(-g));
        float s1 = mainp[pack_in[tid]];
        float s2 = mainp[pack_in[CC + tid]];
        Osc[tid] = (s1 / (1.f + __expf(-s1))) * s2;                     // silu(s1)*s2
    }
    __syncthreads();

    // ---------------- Phase 1: pack-gather into x_arr^T ----------------
    for (int e = tid; e < MAINW; e += 256) {
        int j = e >> 8, c2 = e & 255;
        XT[c2 * JPAD + j] = (_Float16)Raw[pack_in[e]];
    }
    __syncthreads();

    // ---------------- Phase 2: GEMM1 (x_grid) + prod -> PT ----------------
    const int ct   = wave;                          // c-tile pair (ct, ct+8)
    const int cloc = ct * 16 + (lane & 15);
    const int hi8  = (lane >> 4) << 3;
    v16h blo = load_bfrag(XT + (ct * 16)       * JPAD, JPAD, 0, lane);
    v16h bhi = load_bfrag(XT + (128 + ct * 16) * JPAD, JPAD, 0, lane);
    for (int at = 0; at < 13; ++at) {
        v16h a = load_afrag(Tl + (at * 16) * JPAD, JPAD, 0, lane);
        v8f z = {};
        v8f d1 = __builtin_amdgcn_wmma_f32_16x16x32_f16(false, a, false, blo, (short)0, z, false, false);
        v8f d2 = __builtin_amdgcn_wmma_f32_16x16x32_f16(false, a, false, bhi, (short)0, z, false, false);
        int abase = at * 16 + hi8;
        #pragma unroll
        for (int r = 0; r < 8; r += 2) {
            int aidx = abase + r;                   // pairs never straddle a=196
            if (aidx < AA) {
                h2 p;
                p[0] = (_Float16)(d1[r]     * d2[r]);
                p[1] = (_Float16)(d1[r + 1] * d2[r + 1]);
                *(h2*)(PT + cloc * APAD + aidx) = p;
            }
        }
    }
    for (int i = tid; i < 128 * 14; i += 256) {     // zero PT K-tail (a in [196,224))
        int c = i / 14, s = i - c * 14;
        ((unsigned int*)(PT + c * APAD + AA))[s] = 0u;
    }
    __syncthreads();

    // ---------------- Phase 3: GEMM2 (from_grid @ prod) ----------------
    const int cbase = ct * 16;
    v8f acc0 = {}, acc1 = {};
    for (int ks = 0; ks < 7; ++ks) {
        v16h a0 = load_afrag(Fl,             APAD, ks * 32, lane);  // j 0..15
        v16h a1 = load_afrag(Fl + 16 * APAD, APAD, ks * 32, lane);  // j 16..31 (pad zero)
        v16h b  = load_bfrag(PT + cbase * APAD, APAD, ks * 32, lane);
        acc0 = __builtin_amdgcn_wmma_f32_16x16x32_f16(false, a0, false, b, (short)0, acc0, false, false);
        acc1 = __builtin_amdgcn_wmma_f32_16x16x32_f16(false, a1, false, b, (short)0, acc1, false, false);
    }
    {   // gate + swiglu-scalar epilogue into f32 LDS (aliases dead XT)
        const int c  = cbase + (lane & 15);
        const float sg = Sig[c];
        const float ov = Osc[c];
        #pragma unroll
        for (int r = 0; r < 8; ++r) {
            int j0 = r + hi8;                       // 0..15
            OutL[j0 * CC + c] = acc0[r] * sg + ((j0 == 0) ? ov : 0.f);
            int j1 = 16 + j0;                       // 16..31, keep < 25
            if (j1 < JJ) OutL[j1 * CC + c] = acc1[r] * sg;
        }
    }
    __syncthreads();

    // ---------------- Phase 4: unpack-scatter (coalesced global writes) ----------------
    float* outp = out + (size_t)n * OUTROW;
    for (int k = tid; k < OUTROW; k += 256)
        outp[k] = OutL[unpack_out[k]];
}

extern "C" void kernel_launch(void* const* d_in, const int* in_sizes, int n_in,
                              void* d_out, int out_size, void* d_ws, size_t ws_size,
                              hipStream_t stream) {
    const float* x_flat    = (const float*)d_in[0];
    const float* to_grid   = (const float*)d_in[1];
    const float* from_grid = (const float*)d_in[2];
    const int*   pack_in   = (const int*)d_in[3];
    const int*   unpack    = (const int*)d_in[4];
    float* out = (float*)d_out;
    const int n = in_sizes[0] / NROW;
    s2_swiglu_kernel<<<n, 256, 0, stream>>>(x_flat, to_grid, from_grid, pack_in, unpack, out);
}
